// MultiHeadedAttention_16252156248520
// MI455X (gfx1250) — matrix operand order repaired and checked by reading, not compiled, not run on hardware
//
#include <hip/hip_runtime.h>
#include <cstdint>
#include <cstddef>

// Problem constants (match reference)
#define Bb  4
#define Ss  2048
#define Dd  1024
#define Hh  16
#define DHd 64
#define Mm  (Bb * Ss)   // 8192

typedef __bf16        v16bf  __attribute__((ext_vector_type(16)));
typedef float         v8f    __attribute__((ext_vector_type(8)));
typedef unsigned int  u32x4  __attribute__((ext_vector_type(4)));

union TileB {
    unsigned int w[8];
    v16bf        v;
};

// float -> bf16 round-to-nearest-even (inputs are finite; NaN path ignored)
__device__ __forceinline__ unsigned short f2bf(float f) {
    unsigned int u = __builtin_bit_cast(unsigned int, f);
    u += 0x7FFFu + ((u >> 16) & 1u);
    return (unsigned short)(u >> 16);
}

// Two 16B loads -> one 16-element bf16 WMMA operand
__device__ __forceinline__ v16bf ldtile(const unsigned short* p0, const unsigned short* p1) {
    TileB t;
    *(u32x4*)(&t.w[0]) = *(const u32x4*)p0;
    *(u32x4*)(&t.w[4]) = *(const u32x4*)p1;
    return t.v;
}

__device__ __forceinline__ v8f wmma_bf16(v16bf a, v16bf b, v8f c) {
    return __builtin_amdgcn_wmma_f32_16x16x32_bf16(
        /*neg_a=*/false, a, /*neg_b=*/false, b,
        /*c_mod=*/(short)0, c, /*reuse_a=*/false, /*reuse_b=*/false);
}

// ---------------------------------------------------------------------------
// fp32 -> bf16 conversion (grid-stride)
// ---------------------------------------------------------------------------
__global__ void cvt_bf16(const float* __restrict__ in, unsigned short* __restrict__ out, int n) {
    int i = blockIdx.x * blockDim.x + threadIdx.x;
    int stride = gridDim.x * blockDim.x;
    for (; i < n; i += stride) out[i] = f2bf(in[i]);
}

// ---------------------------------------------------------------------------
// GEMM: out[m,n] = sum_k X[m,k]*W[n,k] + bias[n]   (X: [M,1024], W: [1024,1024])
// mode 0: store bf16, layout [B,H,S,DH], scaled by 1/sqrt(DH)   (Q)
// mode 1: store bf16, layout [B,H,S,DH]                          (K)
// mode 2: store bf16, layout [B,H,DH,S] (transposed V)           (V)
// mode 3: store f32 row-major [M,1024]                           (O)
// Each wave computes one 64x64 C macro-tile (4x4 WMMA tiles):
// 16 WMMAs per 16 b128 loads per k-step (4x operand reuse).
// ---------------------------------------------------------------------------
__global__ __launch_bounds__(256) void gemm_xwT(
    const unsigned short* __restrict__ X,
    const unsigned short* __restrict__ W,
    const float* __restrict__ bias,
    void* __restrict__ out, int mode)
{
    const int wave = (blockIdx.x * blockDim.x + threadIdx.x) >> 5;
    const int lane = threadIdx.x & 31;
    const int nl = lane & 15, hi = lane >> 4;
    const int NT = Dd / 64;                 // 16 macro-tiles along N
    const int m0 = (wave / NT) * 64;
    const int n0 = (wave % NT) * 64;

    v8f acc[4][4];
#pragma unroll
    for (int mi = 0; mi < 4; ++mi)
#pragma unroll
        for (int ni = 0; ni < 4; ++ni) acc[mi][ni] = (v8f)(0.f);

    const unsigned short* xr[4];
    const unsigned short* wr[4];
#pragma unroll
    for (int i = 0; i < 4; ++i) {
        xr[i] = X + (size_t)(m0 + i * 16 + nl) * Dd;
        wr[i] = W + (size_t)(n0 + i * 16 + nl) * Dd;
    }

    for (int k0 = 0; k0 < Dd; k0 += 32) {
        v16bf a[4], bt[4];
#pragma unroll
        for (int i = 0; i < 4; ++i)
            a[i] = ldtile(xr[i] + k0 + hi * 8, xr[i] + k0 + 16 + hi * 8);
#pragma unroll
        for (int i = 0; i < 4; ++i)
            bt[i] = ldtile(wr[i] + k0 + hi * 8, wr[i] + k0 + 16 + hi * 8);
        if (k0 + 32 < Dd) {
            __builtin_prefetch(xr[0] + k0 + 32 + hi * 8, 0, 3);
            __builtin_prefetch(wr[0] + k0 + 32 + hi * 16, 0, 3);
        }
#pragma unroll
        for (int mi = 0; mi < 4; ++mi)
#pragma unroll
            for (int ni = 0; ni < 4; ++ni)
                acc[mi][ni] = wmma_bf16(a[mi], bt[ni], acc[mi][ni]);
    }

#pragma unroll
    for (int mi = 0; mi < 4; ++mi) {
#pragma unroll
        for (int ni = 0; ni < 4; ++ni) {
            const int n = n0 + ni * 16 + nl;
            const float bn = bias[n];
            v8f a = acc[mi][ni];
#pragma unroll
            for (int r = 0; r < 8; ++r) {
                const int m = m0 + mi * 16 + hi * 8 + r;   // C row = hi*8+r
                float val = a[r] + bn;
                if (mode == 0) val *= 0.125f;              // 1/sqrt(64)
                if (mode == 3) {
                    ((float*)out)[(size_t)m * Dd + n] = val;
                } else {
                    const int bidx = m >> 11, s = m & (Ss - 1);
                    const int h = n >> 6, d = n & 63;
                    size_t idx;
                    if (mode == 2) idx = (((size_t)bidx * Hh + h) * DHd + d) * Ss + s;
                    else           idx = (((size_t)bidx * Hh + h) * Ss + s) * DHd + d;
                    ((unsigned short*)out)[idx] = f2bf(val);
                }
            }
        }
    }
}

// ---------------------------------------------------------------------------
// Fused flash attention. One wave owns one (b, h, 32-query tile) = two 16-wide
// q-tiles, streaming all 2048 keys in chunks of 32. scores^T via WMMA
// (lane = query), per-lane online softmax, P^T rebuilt as a WMMA B operand
// with shfl_xor(16), ctx^T = V^T x P^T accumulated in fp32 C tiles.
// Each K-tile and V^T-tile load now feeds two WMMAs (one per q-tile).
// ---------------------------------------------------------------------------
__global__ __launch_bounds__(128) void flash_attn(
    const unsigned short* __restrict__ qp,   // [B,H,S,DH] bf16 (pre-scaled)
    const unsigned short* __restrict__ kp,   // [B,H,S,DH] bf16
    const unsigned short* __restrict__ vt,   // [B,H,DH,S] bf16
    const unsigned char*  __restrict__ mask, // [B,S,S] bool
    unsigned short* __restrict__ cbuf)       // [B,S,D] bf16
{
    const int wave = (blockIdx.x * blockDim.x + threadIdx.x) >> 5;
    const int lane = threadIdx.x & 31;
    const int nl = lane & 15, hi = lane >> 4;
    const int QT = Ss / 32;                  // 64 query-chunks per (b,h)
    const int b = wave / (Hh * QT);
    const int rem = wave % (Hh * QT);
    const int h = rem / QT;
    const int q0 = (rem % QT) * 32;

    const unsigned short* qb = qp + (((size_t)b * Hh + h) * Ss + q0) * DHd;
    const unsigned short* kb = kp + (((size_t)b * Hh + h) * Ss) * DHd;
    const unsigned short* vb = vt + (((size_t)b * Hh + h) * DHd) * Ss;
    const unsigned char* mr[2];
    mr[0] = mask + ((size_t)b * Ss + q0 + nl) * Ss;
    mr[1] = mask + ((size_t)b * Ss + q0 + 16 + nl) * Ss;

    // Q^T B-operands, held for the whole key loop: [qtile][d-step]
    v16bf bq[2][2];
#pragma unroll
    for (int qt = 0; qt < 2; ++qt)
#pragma unroll
        for (int st = 0; st < 2; ++st) {
            const unsigned short* p = qb + (qt * 16 + nl) * DHd + st * 32 + hi * 8;
            bq[qt][st] = ldtile(p, p + 16);
        }

    v8f acc[2][4];
#pragma unroll
    for (int qt = 0; qt < 2; ++qt)
#pragma unroll
        for (int dt = 0; dt < 4; ++dt) acc[qt][dt] = (v8f)(0.f);
    float mrun[2] = {-1e30f, -1e30f};
    float lrun[2] = {0.f, 0.f};

    for (int c = 0; c < Ss; c += 32) {
        // --- scores^T tiles: rows = keys, cols = queries. s[qtile][keytile] ---
        v8f s[2][2];
#pragma unroll
        for (int qt = 0; qt < 2; ++qt)
#pragma unroll
            for (int kt = 0; kt < 2; ++kt) s[qt][kt] = (v8f)(0.f);
#pragma unroll
        for (int kt = 0; kt < 2; ++kt) {
            const unsigned short* kr = kb + (size_t)(c + kt * 16 + nl) * DHd;
            v16bf a0 = ldtile(kr + hi * 8, kr + 16 + hi * 8);
            v16bf a1 = ldtile(kr + 32 + hi * 8, kr + 48 + hi * 8);
            s[0][kt] = wmma_bf16(a0, bq[0][0], s[0][kt]);
            s[0][kt] = wmma_bf16(a1, bq[0][1], s[0][kt]);
            s[1][kt] = wmma_bf16(a0, bq[1][0], s[1][kt]);
            s[1][kt] = wmma_bf16(a1, bq[1][1], s[1][kt]);
        }

        TileB bp[2];
#pragma unroll
        for (int qt = 0; qt < 2; ++qt) {
            // --- mask: lane holds query q0+qt*16+nl, keys c + kt*16 + hi*8 + r ---
            const unsigned long long mb0 = *(const unsigned long long*)(mr[qt] + c + hi * 8);
            const unsigned long long mb1 = *(const unsigned long long*)(mr[qt] + c + 16 + hi * 8);
#pragma unroll
            for (int r = 0; r < 8; ++r) {
                if ((mb0 >> (8 * r)) & 0xffull) s[qt][0][r] = -1e18f;
                if ((mb1 >> (8 * r)) & 0xffull) s[qt][1][r] = -1e18f;
            }
            // --- online softmax (per-lane = per-query; partner lane = xor 16) ---
            float cmax = s[qt][0][0];
#pragma unroll
            for (int r = 1; r < 8; ++r) cmax = fmaxf(cmax, s[qt][0][r]);
#pragma unroll
            for (int r = 0; r < 8; ++r) cmax = fmaxf(cmax, s[qt][1][r]);
            cmax = fmaxf(cmax, __shfl_xor(cmax, 16, 32));
            const float mnew = fmaxf(mrun[qt], cmax);
            const float scale = __expf(mrun[qt] - mnew);
            float p0[8], p1[8];
            float lsum = 0.f;
#pragma unroll
            for (int r = 0; r < 8; ++r) { p0[r] = __expf(s[qt][0][r] - mnew); lsum += p0[r]; }
#pragma unroll
            for (int r = 0; r < 8; ++r) { p1[r] = __expf(s[qt][1][r] - mnew); lsum += p1[r]; }
            lsum += __shfl_xor(lsum, 16, 32);
            lrun[qt] = lrun[qt] * scale + lsum;
            mrun[qt] = mnew;
#pragma unroll
            for (int dt = 0; dt < 4; ++dt)
#pragma unroll
                for (int r = 0; r < 8; ++r) acc[qt][dt][r] *= scale;
            // --- build P^T B-operand (32 keys x 16 queries) ---
            unsigned int pk0[4], pk1[4], ot0[4], ot1[4];
#pragma unroll
            for (int j = 0; j < 4; ++j) {
                pk0[j] = (unsigned int)f2bf(p0[2 * j]) | ((unsigned int)f2bf(p0[2 * j + 1]) << 16);
                pk1[j] = (unsigned int)f2bf(p1[2 * j]) | ((unsigned int)f2bf(p1[2 * j + 1]) << 16);
            }
#pragma unroll
            for (int j = 0; j < 4; ++j) {
                ot0[j] = (unsigned int)__shfl_xor((int)pk0[j], 16, 32);
                ot1[j] = (unsigned int)__shfl_xor((int)pk1[j], 16, 32);
            }
#pragma unroll
            for (int j = 0; j < 4; ++j) {
                bp[qt].w[j]     = hi ? ot1[j] : pk0[j];   // k 0..7 (hi=0), 16..23 (hi=1)
                bp[qt].w[4 + j] = hi ? pk1[j] : ot0[j];   // k 8..15 (hi=0), 24..31 (hi=1)
            }
        }

        // --- ctx^T += V^T-tile x P^T : each V^T load feeds both q-tiles ---
#pragma unroll
        for (int dt = 0; dt < 4; ++dt) {
            const unsigned short* vr = vb + (size_t)(dt * 16 + nl) * Ss + c;
            v16bf av = ldtile(vr + hi * 16, vr + hi * 16 + 8);
            acc[0][dt] = wmma_bf16(av, bp[0].v, acc[0][dt]);
            acc[1][dt] = wmma_bf16(av, bp[1].v, acc[1][dt]);
        }
    }

    // --- normalize and store ctx (lane = query; rows of C = d) ---
#pragma unroll
    for (int qt = 0; qt < 2; ++qt) {
        const float rl = 1.f / lrun[qt];
        unsigned short* crow = cbuf + ((size_t)b * Ss + q0 + qt * 16 + nl) * Dd + h * DHd;
#pragma unroll
        for (int dt = 0; dt < 4; ++dt) {
            v8f a = acc[qt][dt];
            unsigned int pk[4];
#pragma unroll
            for (int j = 0; j < 4; ++j)
                pk[j] = (unsigned int)f2bf(a[2 * j] * rl) |
                        ((unsigned int)f2bf(a[2 * j + 1] * rl) << 16);
            *(u32x4*)(crow + dt * 16 + hi * 8) = *(const u32x4*)pk;   // 8 consecutive d
        }
    }
}

// ---------------------------------------------------------------------------
// Host-side launch. Workspace layout (needs ~120 MiB of d_ws):
//   [0,16M)   key bf16        [16M,32M)  value bf16     [32M,48M) query bf16
//   [48M..56M) Wq/Wk/Wv/Wo bf16 (2 MiB each)
//   [56M,72M) qp  [72M,88M) kp  [88M,104M) vt  [104M,120M) ctx bf16
// ---------------------------------------------------------------------------
extern "C" void kernel_launch(void* const* d_in, const int* in_sizes, int n_in,
                              void* d_out, int out_size, void* d_ws, size_t ws_size,
                              hipStream_t stream) {
    const float* key_f = (const float*)d_in[0];
    const float* val_f = (const float*)d_in[1];
    const float* qry_f = (const float*)d_in[2];
    const unsigned char* mask = (const unsigned char*)d_in[3];
    const float* Wq = (const float*)d_in[4];
    const float* bq = (const float*)d_in[5];
    const float* Wk = (const float*)d_in[6];
    const float* bk = (const float*)d_in[7];
    const float* Wv = (const float*)d_in[8];
    const float* bv = (const float*)d_in[9];
    const float* Wo = (const float*)d_in[10];
    const float* bo = (const float*)d_in[11];

    char* ws = (char*)d_ws;
    const size_t MB = 1ull << 20;
    unsigned short* xk  = (unsigned short*)(ws + 0 * MB);
    unsigned short* xv  = (unsigned short*)(ws + 16 * MB);
    unsigned short* xq  = (unsigned short*)(ws + 32 * MB);
    unsigned short* wqb = (unsigned short*)(ws + 48 * MB);
    unsigned short* wkb = (unsigned short*)(ws + 50 * MB);
    unsigned short* wvb = (unsigned short*)(ws + 52 * MB);
    unsigned short* wob = (unsigned short*)(ws + 54 * MB);
    unsigned short* qpw = (unsigned short*)(ws + 56 * MB);
    unsigned short* kpw = (unsigned short*)(ws + 72 * MB);
    unsigned short* vtw = (unsigned short*)(ws + 88 * MB);
    unsigned short* cbw = (unsigned short*)(ws + 104 * MB);

    const int nAct = Mm * Dd;   // 8388608
    const int nW = Dd * Dd;     // 1048576
    cvt_bf16<<<2048, 256, 0, stream>>>(key_f, xk, nAct);
    cvt_bf16<<<2048, 256, 0, stream>>>(val_f, xv, nAct);
    cvt_bf16<<<2048, 256, 0, stream>>>(qry_f, xq, nAct);
    cvt_bf16<<<1024, 256, 0, stream>>>(Wq, wqb, nW);
    cvt_bf16<<<1024, 256, 0, stream>>>(Wk, wkb, nW);
    cvt_bf16<<<1024, 256, 0, stream>>>(Wv, wvb, nW);
    cvt_bf16<<<1024, 256, 0, stream>>>(Wo, wob, nW);

    // 2048 waves = (8192/64) * (1024/64) 64x64 macro-tiles; 8 waves/block.
    gemm_xwT<<<256, 256, 0, stream>>>(xq, wqb, bq, qpw, 0);
    gemm_xwT<<<256, 256, 0, stream>>>(xk, wkb, bk, kpw, 1);
    gemm_xwT<<<256, 256, 0, stream>>>(xv, wvb, bv, vtw, 2);

    // 4096 waves: one per (b, h, 32-query tile); 4 waves/block.
    flash_attn<<<1024, 128, 0, stream>>>(qpw, kpw, vtw, mask, cbw);

    gemm_xwT<<<256, 256, 0, stream>>>(cbw, wob, bo, d_out, 3);
}
